// NodeModelBase_37177236914589
// MI455X (gfx1250) — matrix operand order repaired and checked by reading, not compiled, not run on hardware
//
#include <hip/hip_runtime.h>

#define N_NODES 50000
#define N_EDGES 800000
#define HS      128
#define LN_EPS  1e-5f

// d_ws layout: [0, N_NODES) floats = scatter counts; at byte 204800: packed bf16 weights
#define WS_WOFF 204800

typedef __attribute__((ext_vector_type(16))) __bf16 v16bf;
typedef __attribute__((ext_vector_type(8)))  float  v8f;
typedef __attribute__((ext_vector_type(4)))  unsigned int v4u;
typedef __attribute__((ext_vector_type(8)))  int v8i;
typedef __attribute__((ext_vector_type(4)))  int v4i;

union BfA { v16bf v; unsigned int u[8]; uint4 q[2]; };

// exact RNE pack, used only in the one-shot weight prep kernel
__device__ __forceinline__ unsigned int pk2bf_rne(float a, float b) {
  unsigned int ua = __builtin_bit_cast(unsigned int, a);
  unsigned int ub = __builtin_bit_cast(unsigned int, b);
  ua = (ua + 0x7FFFu + ((ua >> 16) & 1u)) >> 16;
  ub = (ub + 0x7FFFu + ((ub >> 16) & 1u)) >> 16;
  return (ua & 0xFFFFu) | (ub << 16);
}
// fast pack: round-half-up then one v_perm_b32 merging the two high halves
__device__ __forceinline__ unsigned int pk2bf_fast(float a, float b) {
  unsigned int ua = __builtin_bit_cast(unsigned int, a) + 0x8000u;
  unsigned int ub = __builtin_bit_cast(unsigned int, b) + 0x8000u;
  return __builtin_amdgcn_perm(ub, ua, 0x07060302u);
}

// XOR-swizzled ushort offset inside a [row][128] ushort array (16B-block moves:
// keeps ds_load_b128 alignment, kills stride-64-word same-bank column accesses).
__device__ __forceinline__ int swz(int row, int off) {
  return row * 128 + ((((off >> 3) ^ (row & 15)) << 3) | (off & 7));
}

__device__ __forceinline__ void atomAddF(float* p, float v) {
  __hip_atomic_fetch_add(p, v, __ATOMIC_RELAXED, __HIP_MEMORY_SCOPE_AGENT);
}

__global__ void zero_kernel(float* __restrict__ out, float* __restrict__ cnt) {
  int i = blockIdx.x * blockDim.x + threadIdx.x;
  if (i < N_NODES * HS) out[i] = 0.0f;
  if (i < N_NODES)      cnt[i] = 0.0f;
}

__global__ void count_kernel(const int* __restrict__ col, float* __restrict__ cnt) {
  int e = blockIdx.x * blockDim.x + threadIdx.x;
  if (e < N_EDGES) atomAddF(&cnt[col[e]], 1.0f);
}

__global__ void finalize_kernel(float* __restrict__ out, const float* __restrict__ cnt) {
  int i = blockIdx.x * blockDim.x + threadIdx.x;
  if (i < N_NODES * HS) {
    float c = cnt[i >> 7];
    out[i] = out[i] / fmaxf(c, 1.0f);
  }
}

// One-shot: W1ᵀ/W2ᵀ -> bf16 (RNE), written as the exact linear image of the
// swizzled LDS buffers, so the hot kernel stages with one TDM load per matrix.
__global__ void pack_weights(const float* __restrict__ W1, const float* __restrict__ W2,
                             unsigned int* __restrict__ wsw) {
  int t = blockIdx.x * blockDim.x + threadIdx.x;     // 0..8191
  if (t >= 8192) return;
  const int m    = t >> 12;          // 0: W1, 1: W2
  const int nn   = t & 127;          // output feature = row of Wᵀ
  const int kblk = (t >> 7) & 31;    // input-feature quad
  const int k0   = kblk * 4;
  const float* W = m ? W2 : W1;
  unsigned int d0 = pk2bf_rne(W[(k0 + 0) * HS + nn], W[(k0 + 1) * HS + nn]);
  unsigned int d1 = pk2bf_rne(W[(k0 + 2) * HS + nn], W[(k0 + 3) * HS + nn]);
  unsigned int* p = wsw + m * 8192 + (swz(nn, k0) >> 1);
  p[0] = d0; p[1] = d1;
}

// Issue one TDM tensor_load_to_lds moving `dwords` dwords from `gsrc` to LDS
// offset `lds_off`.  D# built per CDNA5 ISA 08_async_tensor §8 (2D, count=1,
// data_size=4B, type=2 "image").  This toolchain carries the 6-arg builtin
// (g0, g1, g2, g3, extra v8i group, cpol) — trailing groups zero for 2D.
#if __has_builtin(__builtin_amdgcn_tensor_load_to_lds)
#define HAVE_TDM 1
__device__ __forceinline__ void tdm_load(const void* gsrc, unsigned int lds_off,
                                         unsigned int dwords) {
  unsigned long long ga = (unsigned long long)(uintptr_t)gsrc;
  v4u g0;
  g0[0] = 1u;                                        // count=1, user mode, no gather
  g0[1] = lds_off;                                   // lds_addr
  g0[2] = (unsigned int)ga;                          // global_addr[31:0]
  g0[3] = (unsigned int)((ga >> 32) & 0x1FFFFFFu)    // global_addr[56:32]
        | (2u << 30);                                // type=2
  v8i g1;
  g1[0] = (int)(2u << 16);                           // data_size=2 (4 bytes), mask=0
  g1[1] = (int)((dwords & 0xFFFFu) << 16);           // tensor_dim0[15:0]
  g1[2] = (int)(((dwords >> 16) & 0xFFFFu)           // tensor_dim0[31:16]
        | (1u << 16));                               // tensor_dim1 = 1
  g1[3] = (int)((dwords & 0xFFFFu) << 16);           // tile_dim0
  g1[4] = 1;                                         // tile_dim1 = 1, tile_dim2 = 0
  g1[5] = (int)dwords;                               // tensor_dim0_stride[31:0]
  g1[6] = 0;                                         // stride[47:32], dim1_stride lo
  g1[7] = 0;
  v4i gz4 = {0, 0, 0, 0};
  v8i gz8 = {0, 0, 0, 0, 0, 0, 0, 0};
  __builtin_amdgcn_tensor_load_to_lds(g0, g1, gz4, gz4, gz8, 0);
}
#endif

// Transposed formulation:  Hᵀ(128f x 16e) = W1ᵀ·Xᵀ ;  Yᵀ = W2ᵀ·relu(LN(Hᵀ)).
// A = weight tiles (LDS, A-layout), B = activations (lane = edge column).
__global__ __launch_bounds__(256)
void edge_mlp_wmma(const float* __restrict__ x,  const float* __restrict__ ea,
                   const int*   __restrict__ rowi, const int* __restrict__ coli,
                   const unsigned int* __restrict__ wsw,
                   const float* __restrict__ b1,
                   const float* __restrict__ lng, const float* __restrict__ lnb,
                   const float* __restrict__ b2,
                   float* __restrict__ out) {
  __shared__ __align__(16) unsigned short lds_w1[HS * HS];   // W1ᵀ bf16, swizzled
  __shared__ __align__(16) unsigned short lds_w2[HS * HS];   // W2ᵀ bf16, swizzled

  const int tid  = threadIdx.x;
  const int wave = tid >> 5;
  const int lane = tid & 31;
  const int n    = lane & 15;   // edge column (B/C/D) / row-in-tile (A)
  const int hf   = lane >> 4;   // wave half

  // ---------------- stage pre-packed weights into LDS ----------------
#ifdef HAVE_TDM
  if (wave == 0) {
    tdm_load(wsw,        (unsigned int)(uintptr_t)&lds_w1[0], 8192u);
    tdm_load(wsw + 8192, (unsigned int)(uintptr_t)&lds_w2[0], 8192u);
#if __has_builtin(__builtin_amdgcn_s_wait_tensorcnt)
    __builtin_amdgcn_s_wait_tensorcnt(0);
#else
    asm volatile("s_wait_tensorcnt 0x0" ::: "memory");
#endif
  }
#else
  {
    const uint4* ws4 = (const uint4*)wsw;
    uint4* l1 = (uint4*)lds_w1;
    uint4* l2 = (uint4*)lds_w2;
    #pragma unroll
    for (int i = 0; i < 8; ++i) l1[i * 256 + tid] = ws4[i * 256 + tid];
    #pragma unroll
    for (int i = 0; i < 8; ++i) l2[i * 256 + tid] = ws4[2048 + i * 256 + tid];
  }
#endif
  __syncthreads();

  // ---------------- B1 = Xᵀ chunk: lane = edge, 16 consecutive features per kc ----------------
  const int ebase = blockIdx.x * 128 + wave * 16;
  const int e     = ebase + n;
  const int xrow  = rowi[e];

  unsigned int bx[4][8];
  #pragma unroll
  for (int kc = 0; kc < 4; ++kc) {
    const int f0 = kc * 32 + hf * 16;                    // K range of this half
    const float* src = (kc < 2) ? (x  + (size_t)xrow * 64 + f0)
                                : (ea + (size_t)e    * 64 + (f0 - 64));
    float4 g0 = *(const float4*)(src);
    float4 g1 = *(const float4*)(src + 4);
    float4 g2 = *(const float4*)(src + 8);
    float4 g3 = *(const float4*)(src + 12);
    bx[kc][0] = pk2bf_fast(g0.x, g0.y); bx[kc][1] = pk2bf_fast(g0.z, g0.w);
    bx[kc][2] = pk2bf_fast(g1.x, g1.y); bx[kc][3] = pk2bf_fast(g1.z, g1.w);
    bx[kc][4] = pk2bf_fast(g2.x, g2.y); bx[kc][5] = pk2bf_fast(g2.z, g2.w);
    bx[kc][6] = pk2bf_fast(g3.x, g3.y); bx[kc][7] = pk2bf_fast(g3.z, g3.w);
  }

  // ---------------- GEMM1: acc[mt] = W1ᵀ(16x32) x Xᵀ(32x16), init = b1 ----------------
  v8f acc[8];
  #pragma unroll
  for (int mt = 0; mt < 8; ++mt) {                       // features mt*16 + 8*hf + v
    const float* bp = b1 + mt * 16 + 8 * hf;
    float4 c0 = *(const float4*)bp;
    float4 c1 = *(const float4*)(bp + 4);
    acc[mt][0] = c0.x; acc[mt][1] = c0.y; acc[mt][2] = c0.z; acc[mt][3] = c0.w;
    acc[mt][4] = c1.x; acc[mt][5] = c1.y; acc[mt][6] = c1.z; acc[mt][7] = c1.w;
  }
  #pragma unroll
  for (int kc = 0; kc < 4; ++kc) {
    BfA b;
    #pragma unroll
    for (int p = 0; p < 8; ++p) b.u[p] = bx[kc][p];
    #pragma unroll
    for (int mt = 0; mt < 8; ++mt) {
      BfA a;
      const int r = mt * 16 + n;
      const int o = kc * 32 + hf * 16;                   // contiguous K half split
      a.q[0] = *(const uint4*)&lds_w1[swz(r, o)];
      a.q[1] = *(const uint4*)&lds_w1[swz(r, o + 8)];
      acc[mt] = __builtin_amdgcn_wmma_f32_16x16x32_bf16(
          false, a.v, false, b.v, (short)0, acc[mt], false, false);
    }
  }

  // ---------------- LayerNorm per edge (per lane column): one half-swap shuffle ----------------
  float s = 0.f, q = 0.f;
  #pragma unroll
  for (int mt = 0; mt < 8; ++mt)
    #pragma unroll
    for (int v = 0; v < 8; ++v) { float t = acc[mt][v]; s += t; q += t * t; }
  s += __shfl_xor(s, 16, 32);
  q += __shfl_xor(q, 16, 32);
  const float mu     = s * (1.0f / 128.0f);
  const float inv    = rsqrtf(fmaxf(q * (1.0f / 128.0f) - mu * mu, 0.0f) + LN_EPS);
  const float nmuinv = -mu * inv;

  // normalize + affine + relu, pack bf16 pairs (3 VALU ops + 1 perm per pair)
  unsigned int hn[8][4];
  #pragma unroll
  for (int mt = 0; mt < 8; ++mt) {
    const float* gp = lng + mt * 16 + 8 * hf;
    const float* bp = lnb + mt * 16 + 8 * hf;
    float4 g0 = *(const float4*)gp, g1 = *(const float4*)(gp + 4);
    float4 o0 = *(const float4*)bp, o1 = *(const float4*)(bp + 4);
    float gv[8] = {g0.x, g0.y, g0.z, g0.w, g1.x, g1.y, g1.z, g1.w};
    float ov[8] = {o0.x, o0.y, o0.z, o0.w, o1.x, o1.y, o1.z, o1.w};
    float val[8];
    #pragma unroll
    for (int v = 0; v < 8; ++v) {
      float t = __builtin_fmaf(acc[mt][v], inv, nmuinv);
      val[v]  = fmaxf(__builtin_fmaf(t, gv[v], ov[v]), 0.0f);
    }
    #pragma unroll
    for (int p = 0; p < 4; ++p) hn[mt][p] = pk2bf_fast(val[2 * p], val[2 * p + 1]);
  }

  // ---------------- GEMM2: acc2[mt] = W2ᵀ x Hᵀ, B built via half-swap of packed pairs ----------------
  v8f acc2[8];
  #pragma unroll
  for (int mt = 0; mt < 8; ++mt) {
    const float* bp = b2 + mt * 16 + 8 * hf;
    float4 c0 = *(const float4*)bp;
    float4 c1 = *(const float4*)(bp + 4);
    acc2[mt][0] = c0.x; acc2[mt][1] = c0.y; acc2[mt][2] = c0.z; acc2[mt][3] = c0.w;
    acc2[mt][4] = c1.x; acc2[mt][5] = c1.y; acc2[mt][6] = c1.z; acc2[mt][7] = c1.w;
  }
  #pragma unroll
  for (int kc = 0; kc < 4; ++kc) {
    // K chunk kc covers H-features kc*32..+31 = tiles 2kc (hf0 K0-15) / 2kc+1 (hf1 K16-31).
    unsigned int loc[4], rem[4];
    #pragma unroll
    for (int p = 0; p < 4; ++p) {
      unsigned int t = hf ? hn[2 * kc][p] : hn[2 * kc + 1][p];   // what partner needs
      rem[p] = __shfl_xor(t, 16, 32);                            // what I need from partner
      loc[p] = hf ? hn[2 * kc + 1][p] : hn[2 * kc][p];
    }
    BfA b;
    #pragma unroll
    for (int p = 0; p < 4; ++p) {
      b.u[p]     = hf ? rem[p] : loc[p];   // K 0-7  of this half's range
      b.u[p + 4] = hf ? loc[p] : rem[p];   // K 8-15 of this half's range
    }
    #pragma unroll
    for (int mt = 0; mt < 8; ++mt) {
      BfA a;
      const int r = mt * 16 + n;
      const int o = kc * 32 + hf * 16;
      a.q[0] = *(const uint4*)&lds_w2[swz(r, o)];
      a.q[1] = *(const uint4*)&lds_w2[swz(r, o + 8)];
      acc2[mt] = __builtin_amdgcn_wmma_f32_16x16x32_bf16(
          false, a.v, false, b.v, (short)0, acc2[mt], false, false);
    }
  }

  // ---------------- scatter-add: one destination row per lane, contiguous features ----------------
  const int dst = coli[e];
  float* orow = out + (size_t)dst * HS;
  #pragma unroll
  for (int mt = 0; mt < 8; ++mt) {
    const int fb = mt * 16 + 8 * hf;
    #pragma unroll
    for (int v = 0; v < 8; ++v)
      atomAddF(orow + fb + v, acc2[mt][v]);
  }
}

extern "C" void kernel_launch(void* const* d_in, const int* in_sizes, int n_in,
                              void* d_out, int out_size, void* d_ws, size_t ws_size,
                              hipStream_t stream) {
  const float* x    = (const float*)d_in[0];
  const float* ea   = (const float*)d_in[1];
  const int*   ei   = (const int*)  d_in[2];   // [2, N_EDGES] int32
  const float* W1   = (const float*)d_in[3];
  const float* b1   = (const float*)d_in[4];
  const float* lng  = (const float*)d_in[5];
  const float* lnb  = (const float*)d_in[6];
  const float* W2   = (const float*)d_in[7];
  const float* b2   = (const float*)d_in[8];
  float* out = (float*)d_out;
  float* cnt = (float*)d_ws;                                   // N_NODES floats
  unsigned int* wsw = (unsigned int*)((char*)d_ws + WS_WOFF);  // 64 KB packed weights

  const int* rowi = ei;
  const int* coli = ei + N_EDGES;

  zero_kernel<<<(N_NODES * HS + 255) / 256, 256, 0, stream>>>(out, cnt);
  pack_weights<<<32, 256, 0, stream>>>(W1, W2, wsw);
  count_kernel<<<(N_EDGES + 255) / 256, 256, 0, stream>>>(coli, cnt);
  edge_mlp_wmma<<<N_EDGES / 128, 256, 0, stream>>>(x, ea, rowi, coli, wsw,
                                                   b1, lng, lnb, b2, out);
  finalize_kernel<<<(N_NODES * HS + 255) / 256, 256, 0, stream>>>(out, cnt);
}
